// PatchModelPosEnc_48163763257495
// MI455X (gfx1250) — matrix operand order repaired and checked by reading, not compiled, not run on hardware
//
#include <hip/hip_runtime.h>
#include <hip/hip_bf16.h>

typedef _Float16 half_t;
typedef __attribute__((ext_vector_type(8)))  _Float16 v8h;
typedef __attribute__((ext_vector_type(16))) _Float16 v16h;
typedef __attribute__((ext_vector_type(8)))  float    v8f;

#define S0 128
#define S1 32
#define A_SCAN 496
#define COUT 32
#define KH 5
#define KW 9
#define WP 488          // 496 - 9 + 1
#define FOUT 512
#define KPAD 512        // K padded to multiple of 32
#define GM (S0*S1*COUT) // 131072

// ---------------------------------------------------------------------------
// Kernel 0: fc_w [512,488] f32 -> B [512,512] f16, zero-padded K
// ---------------------------------------------------------------------------
__global__ __launch_bounds__(256) void fcw_convert_kernel(
    const float* __restrict__ fcw, half_t* __restrict__ Bout) {
  int idx = blockIdx.x * 256 + threadIdx.x;   // 0 .. 512*512-1
  int n = idx >> 9;
  int k = idx & 511;
  float v = (k < WP) ? fcw[n * WP + k] : 0.0f;
  Bout[idx] = (half_t)v;
}

// ---------------------------------------------------------------------------
// Kernel 1: fused conv (signal channel + analytic coord channel + bias)
// one block per (t, b); writes A[M=131072, 512] f16, K zero-padded
// ---------------------------------------------------------------------------
__global__ __launch_bounds__(256) void conv_kernel(
    const float* __restrict__ seq, const float* __restrict__ cw,
    const float* __restrict__ cb, half_t* __restrict__ Aout) {
  const int t = blockIdx.x;   // 0..127
  const int b = blockIdx.y;   // 0..31
  __shared__ float s_seq[KH * A_SCAN];       // 5 padded-S0 rows of the scan
  __shared__ float s_w[COUT * 2 * KH * KW];  // 2880 weights
  __shared__ float s_b[COUT];

  const int tid = threadIdx.x;
  for (int i = tid; i < COUT * 2 * KH * KW; i += 256) s_w[i] = cw[i];
  if (tid < COUT) s_b[tid] = cb[tid];
  for (int i = tid; i < KH * A_SCAN; i += 256) {
    int kh = i / A_SCAN;
    int a  = i - kh * A_SCAN;
    int ts = t + kh - 2;                     // padded-S0 source row
    s_seq[i] = ((unsigned)ts < (unsigned)S0)
                   ? seq[((size_t)ts * S1 + b) * A_SCAN + a] : 0.0f;
  }
  __syncthreads();

  bool valid[KH];
#pragma unroll
  for (int kh = 0; kh < KH; ++kh) valid[kh] = ((unsigned)(t + kh - 2) < (unsigned)S0);

  const float inv = 1.0f / 248.0f;
  const size_t outbase = (size_t)((t * S1 + b) * COUT) * KPAD;

  for (int idx = tid; idx < COUT * KPAD; idx += 256) {
    int c = idx >> 9;
    int w = idx & 511;
    float sum = 0.0f;
    if (w < WP) {
      sum = s_b[c];
#pragma unroll
      for (int kh = 0; kh < KH; ++kh) {
        if (valid[kh]) {
          const float* w0 = s_w + ((c * 2 + 0) * KH + kh) * KW;
          const float* w1 = s_w + ((c * 2 + 1) * KH + kh) * KW;
          const float* sr = s_seq + kh * A_SCAN + w;
#pragma unroll
          for (int kw = 0; kw < KW; ++kw) {
            float coord = (float)(w + kw - 248) * inv;   // cc 'center' pos-enc
            sum = fmaf(w0[kw], sr[kw], sum);
            sum = fmaf(w1[kw], coord, sum);
          }
        }
      }
    }
    Aout[outbase + idx] = (half_t)sum;   // idx == c*512 + w
  }
}

// ---------------------------------------------------------------------------
// Kernel 2: WMMA GEMM  C[M,512] = A[M,512](f16) * B[512,512](f16,[N,K])^T + bias
// Double-buffered LDS, async global->LDS staging (ASYNCcnt), NT output stores.
// ---------------------------------------------------------------------------
#define BM 128
#define BN 128
#define BK 32
#define BKP 40   // LDS row stride in halves (80B, multiple of 16B, bank-spread)
#define NSTEP (KPAD / BK)   // 16

// 16B global -> LDS async copy (GLOBAL_LOAD_ASYNC_TO_LDS_B128, GV mode).
// LDS hardware address = low 32 bits of the generic shared pointer
// (ISA: LDS aperture maps addr[31:0] to LDS).
__device__ __forceinline__ void async_copy16(const half_t* g, half_t* l) {
  unsigned           lds = (unsigned)(unsigned long long)l;
  unsigned long long ga  = (unsigned long long)g;
  asm volatile("global_load_async_to_lds_b128 %0, %1, off"
               :: "v"(lds), "v"(ga) : "memory");
}
__device__ __forceinline__ void wait_async0() {
  asm volatile("s_wait_asynccnt 0x0" ::: "memory");
}

// A fragment per ISA 16-bit A 16x32 layout:
// lanes 0-15 : halves 0-7 = K0-7,  halves 8-15 = K16-23
// lanes 16-31: halves 0-7 = K8-15, halves 8-15 = K24-31
__device__ __forceinline__ v16h load_frag_a(const half_t* base, int lane) {
  int m = lane & 15;
  int koff = (lane < 16) ? 0 : 8;
  const half_t* p = base + m * BKP + koff;
  v8h lo = *(const v8h*)(p);
  v8h hi = *(const v8h*)(p + 16);
  return __builtin_shufflevector(lo, hi, 0,1,2,3,4,5,6,7,8,9,10,11,12,13,14,15);
}

// B fragment (dense 16-bit B 32x16, column n = lane&15):
// lanes 0-15 : K0-7, K16-23 ; lanes 16-31: K8-15, K24-31
__device__ __forceinline__ v16h load_frag_b(const half_t* base, int lane) {
  int n = lane & 15;
  int kbase = (lane < 16) ? 0 : 8;
  const half_t* p = base + n * BKP + kbase;
  v8h lo = *(const v8h*)(p);
  v8h hi = *(const v8h*)(p + 16);
  return __builtin_shufflevector(lo, hi, 0,1,2,3,4,5,6,7,8,9,10,11,12,13,14,15);
}

__device__ __forceinline__ void stage_async(
    half_t* sa, half_t* sb,
    const half_t* __restrict__ A, const half_t* __restrict__ B,
    int M0, int N0, int k0, int tid) {
#pragma unroll
  for (int it = 0; it < 2; ++it) {
    int idx = tid + it * 256;       // 0..511
    int row = idx >> 2;             // 0..127
    int col = (idx & 3) * 8;        // 0,8,16,24 halves
    async_copy16(A + (size_t)(M0 + row) * KPAD + k0 + col, sa + row * BKP + col);
    async_copy16(B + (size_t)(N0 + row) * KPAD + k0 + col, sb + row * BKP + col);
  }
}

__global__ __launch_bounds__(256) void gemm_kernel(
    const half_t* __restrict__ A, const half_t* __restrict__ B,
    const float* __restrict__ bias, float* __restrict__ C) {
  __shared__ half_t sA[2][BM * BKP];
  __shared__ half_t sB[2][BN * BKP];

  const int tid  = threadIdx.x;
  const int lane = tid & 31;
  const int wave = tid >> 5;      // 0..7
  const int wm   = wave >> 1;     // 0..3 -> M offset wm*32
  const int wn   = wave & 1;      // 0..1 -> N offset wn*64
  const int M0   = blockIdx.y * BM;
  const int N0   = blockIdx.x * BN;   // N-fastest grid: A slab stays in L2

  const v8f vzero = {0.f, 0.f, 0.f, 0.f, 0.f, 0.f, 0.f, 0.f};
  v8f acc[2][4];
#pragma unroll
  for (int i = 0; i < 2; ++i)
#pragma unroll
    for (int j = 0; j < 4; ++j) acc[i][j] = vzero;

  // prologue: fill buffer 0
  stage_async(sA[0], sB[0], A, B, M0, N0, 0, tid);
  wait_async0();
  __syncthreads();

  for (int ks = 0; ks < NSTEP; ++ks) {
    const int cur = ks & 1;
    // issue next tile into the idle buffer (read of it finished last barrier)
    if (ks + 1 < NSTEP)
      stage_async(sA[cur ^ 1], sB[cur ^ 1], A, B, M0, N0, (ks + 1) * BK, tid);

    v16h af[2], bf[4];
#pragma unroll
    for (int i = 0; i < 2; ++i)
      af[i] = load_frag_a(sA[cur] + (wm * 32 + i * 16) * BKP, lane);
#pragma unroll
    for (int j = 0; j < 4; ++j)
      bf[j] = load_frag_b(sB[cur] + (wn * 64 + j * 16) * BKP, lane);

#pragma unroll
    for (int i = 0; i < 2; ++i)
#pragma unroll
      for (int j = 0; j < 4; ++j)
        acc[i][j] = __builtin_amdgcn_wmma_f32_16x16x32_f16(
            false, af[i], false, bf[j], (short)0, acc[i][j], false, false);

    wait_async0();     // own async transfers for ks+1 have landed in LDS
    __syncthreads();   // everyone's have; cur buffer fully consumed
  }

  // epilogue: + fc_b, NT store f32 per C/D layout (VGPR r -> m = r + 8*(lane>=16))
#pragma unroll
  for (int j = 0; j < 4; ++j) {
    int n = N0 + wn * 64 + j * 16 + (lane & 15);
    float bv = bias[n];
#pragma unroll
    for (int i = 0; i < 2; ++i) {
      int mbase = M0 + wm * 32 + i * 16 + ((lane & 16) ? 8 : 0);
#pragma unroll
      for (int r = 0; r < 8; ++r)
        __builtin_nontemporal_store(acc[i][j][r] + bv,
                                    &C[(size_t)(mbase + r) * FOUT + n]);
    }
  }
}

// ---------------------------------------------------------------------------
extern "C" void kernel_launch(void* const* d_in, const int* in_sizes, int n_in,
                              void* d_out, int out_size, void* d_ws, size_t ws_size,
                              hipStream_t stream) {
  const float* seq    = (const float*)d_in[0];
  const float* conv_w = (const float*)d_in[1];
  const float* conv_b = (const float*)d_in[2];
  const float* fc_w   = (const float*)d_in[3];
  const float* fc_b   = (const float*)d_in[4];
  float* out = (float*)d_out;

  half_t* Aw = (half_t*)d_ws;                                       // 128 MB
  half_t* Bw = (half_t*)((char*)d_ws + (size_t)GM * KPAD * sizeof(half_t));

  fcw_convert_kernel<<<(FOUT * KPAD) / 256, 256, 0, stream>>>(fc_w, Bw);
  conv_kernel<<<dim3(S0, S1), 256, 0, stream>>>(seq, conv_w, conv_b, Aw);
  gemm_kernel<<<dim3(FOUT / BN, GM / BM), 256, 0, stream>>>(Aw, Bw, fc_b, out);
}
